// GNN_GAT_71992241815727
// MI455X (gfx1250) — matrix operand order repaired and checked by reading, not compiled, not run on hardware
//
#include <hip/hip_runtime.h>
#include <hip/hip_bf16.h>
#include <math.h>

// ---------------------------------------------------------------------------
// GNN_GAT on gfx1250: WMMA (f16 in / f32 acc) for the node GEMMs with
// async-LDS W staging + swizzled B fragments, atomics for the edge softmax /
// aggregation, single-block MLP head.
// ---------------------------------------------------------------------------

typedef __attribute__((ext_vector_type(16))) _Float16 v16h;
typedef __attribute__((ext_vector_type(8)))  float    v8f;

#define HID 128
#define HEADS 4
#define OUT_CH 32
#define N_GRAPHS 64

#if defined(__gfx1250__) && __has_builtin(__builtin_amdgcn_global_load_async_to_lds_b128)
#define USE_ASYNC_LDS 1
typedef __attribute__((ext_vector_type(4))) int v4i_t;
typedef __attribute__((address_space(1))) v4i_t g_v4i;  // global (device) int4*
typedef __attribute__((address_space(3))) v4i_t l_v4i;  // LDS (shared) int4*
#endif

// Swizzled LDS index so a B fragment (32x16 f16 tile of W) is one contiguous
// 32B v16h per lane:  lane = ((k&8)<<1)|(n&15), element-in-lane = (k&7)|((k&16)>>1).
__device__ __forceinline__ int swz(int k, int n, int KP32) {
  int tile = (n >> 4) * KP32 + (k >> 5);
  int lane = ((k & 8) << 1) | (n & 15);
  return (tile * 32 + lane) * 16 + ((k & 7) | ((k & 16) >> 1));
}

// ---------------------------------------------------------------------------
// H[N x 128] = X[N x K] @ W[K x 128]   (no bias: GAT adds bias post-aggregate)
// 256 threads = 8 waves; wave w owns node rows [blk*128+16w, +16).
// OOB rows are clamped to row N-1 (valid memory) and computed but not stored,
// so A-fragment loads need no guards at all.
// ---------------------------------------------------------------------------
template <int K>
__global__ void gat_linear_wmma(const float* __restrict__ X,
                                const float* __restrict__ W,
                                float* __restrict__ H, int N) {
  constexpr int KP = (K < 32) ? 32 : K;  // K padded to a multiple of 32
  constexpr int KP32 = KP / 32;
  constexpr int CROWS = 32;              // W rows staged per chunk (16 KB f32)
  __shared__ __attribute__((aligned(32))) _Float16 wswz[KP * HID];
  __shared__ __attribute__((aligned(16))) float    wraw[CROWS * HID];

  const int tid = threadIdx.x;
  const int wave = tid >> 5;
  const int lane = tid & 31;
  const int m0 = blockIdx.x * 128 + wave * 16;
  const int mrow0 = m0 + (lane & 15);
  const int mclamp = (mrow0 < N) ? mrow0 : (N - 1);
  const float* xrow = X + (size_t)mclamp * K;

  // warm caches with this wave's A rows while the block stages W
  __builtin_prefetch(xrow, 0, 1);
  if (K > 64) __builtin_prefetch(xrow + 64, 0, 1);

  // zero-fill padded K rows (layer 1: rows 16..31)
  if (KP > K) {
    for (int i = tid; i < (KP - K) * HID; i += 256) {
      int k = K + (i >> 7), n = i & 127;
      wswz[swz(k, n, KP32)] = (_Float16)0.0f;
    }
  }

  // stage W: global --(async DMA)--> LDS raw f32 --> swizzled f16
  for (int k0 = 0; k0 < K; k0 += CROWS) {
    const int rows = (K - k0 < CROWS) ? (K - k0) : CROWS;
    const int elems = rows * HID;  // multiple of 512 -> 16B aligned chunks
#if defined(USE_ASYNC_LDS)
    for (int i = tid * 4; i < elems; i += 256 * 4) {
      __builtin_amdgcn_global_load_async_to_lds_b128(
          (g_v4i*)(const void*)(W + (size_t)k0 * HID + i),
          (l_v4i*)(void*)(wraw + i), 0, 0);
    }
#if __has_builtin(__builtin_amdgcn_s_wait_asynccnt)
    __builtin_amdgcn_s_wait_asynccnt(0);
#else
    asm volatile("s_wait_asynccnt 0" ::: "memory");
#endif
#else
    for (int i = tid; i < elems; i += 256)
      wraw[i] = W[(size_t)k0 * HID + i];
#endif
    __syncthreads();
    for (int i = tid; i < elems; i += 256) {
      int k = k0 + (i >> 7), n = i & 127;
      wswz[swz(k, n, KP32)] = (_Float16)wraw[i];
    }
    __syncthreads();
  }

  if (m0 >= N) return;  // wave-uniform -> EXEC all-1s for every WMMA below
  const bool full = (m0 + 16 <= N);
  const int khalfA = (lane >> 4) * 8;  // A: lanes 0-15 low K-half, 16-31 high
  const int nlane = lane & 15;         // B/C/D column index
  const int hi8 = (lane >> 4) * 8;     // C/D row offset for lanes 16-31

  for (int nt = 0; nt < 8; ++nt) {
    v8f acc = {};
#pragma unroll
    for (int kt32 = 0; kt32 < KP32; ++kt32) {
      const int kt = kt32 * 32;
      // A fragment: unconditional 16B vector loads (row clamped -> safe).
      // Lower K-half of the tile: k = kt+khalfA .. +7  (always < K)
      v16h a;
      {
        const float4* pl = (const float4*)(xrow + kt + khalfA);
        float4 l0 = pl[0], l1 = pl[1];
        a[0] = (_Float16)l0.x; a[1] = (_Float16)l0.y;
        a[2] = (_Float16)l0.z; a[3] = (_Float16)l0.w;
        a[4] = (_Float16)l1.x; a[5] = (_Float16)l1.y;
        a[6] = (_Float16)l1.z; a[7] = (_Float16)l1.w;
      }
      if (K == 16) {
        // Upper K-half (k >= 16) is zero padding for layer 1
#pragma unroll
        for (int j = 8; j < 16; ++j) a[j] = (_Float16)0.0f;
      } else {
        const float4* pu = (const float4*)(xrow + kt + 16 + khalfA);
        float4 u0 = pu[0], u1 = pu[1];
        a[8] = (_Float16)u0.x;  a[9] = (_Float16)u0.y;
        a[10] = (_Float16)u0.z; a[11] = (_Float16)u0.w;
        a[12] = (_Float16)u1.x; a[13] = (_Float16)u1.y;
        a[14] = (_Float16)u1.z; a[15] = (_Float16)u1.w;
      }
      // B fragment: one contiguous 32B LDS load per lane
      const int tile = nt * KP32 + kt32;
      v16h bf = *(const v16h*)(wswz + (tile * 32 + lane) * 16);
      acc = __builtin_amdgcn_wmma_f32_16x16x32_f16(false, a, false, bf,
                                                   (short)0, acc, false, false);
    }
    const int n0 = nt * 16;
    if (full) {
#pragma unroll
      for (int v = 0; v < 8; ++v)
        H[(size_t)(m0 + v + hi8) * HID + n0 + nlane] = acc[v];
    } else {
#pragma unroll
      for (int v = 0; v < 8; ++v) {
        int m = m0 + v + hi8;
        if (m < N) H[(size_t)m * HID + n0 + nlane] = acc[v];
      }
    }
  }
}

// a_s[n][h] = sum_c H[n][h*32+c]*att_src[h][c]; same for a_d
__global__ void attn_scores(const float* __restrict__ H,
                            const float* __restrict__ att_src,
                            const float* __restrict__ att_dst,
                            float* __restrict__ a_s, float* __restrict__ a_d,
                            int N) {
  int i = blockIdx.x * blockDim.x + threadIdx.x;
  if (i >= N * HEADS) return;
  int n = i >> 2, h = i & 3;
  const float* hp = H + (size_t)n * HID + h * OUT_CH;
  const float* as = att_src + h * OUT_CH;
  const float* ad = att_dst + h * OUT_CH;
  float s = 0.f, d = 0.f;
#pragma unroll 4
  for (int c = 0; c < OUT_CH; ++c) {
    float v = hp[c];
    s += v * as[c];
    d += v * ad[c];
  }
  a_s[i] = s;
  a_d[i] = d;
}

// order-preserving float<->uint encoding for atomicMax on floats
__device__ __forceinline__ unsigned fenc(float f) {
  unsigned u = __float_as_uint(f);
  return (u & 0x80000000u) ? ~u : (u | 0x80000000u);
}
__device__ __forceinline__ float fdec(unsigned e) {
  unsigned u = (e & 0x80000000u) ? (e ^ 0x80000000u) : ~e;
  return __uint_as_float(u);
}

// e = leaky_relu(a_s[src]+a_d[dst]); segment max per (dst, head)
__global__ void edge_scores_max(const int* __restrict__ src,
                                const int* __restrict__ dst,
                                const float* __restrict__ a_s,
                                const float* __restrict__ a_d,
                                float* __restrict__ ew,
                                unsigned* __restrict__ menc, int E) {
  int e = blockIdx.x * blockDim.x + threadIdx.x;
  if (e >= E) return;
  int s = src[e], d = dst[e];
#pragma unroll
  for (int h = 0; h < HEADS; ++h) {
    float v = a_s[s * HEADS + h] + a_d[d * HEADS + h];
    v = (v > 0.f) ? v : 0.2f * v;
    ew[(size_t)e * HEADS + h] = v;
    atomicMax(&menc[d * HEADS + h], fenc(v));
  }
}

// w = exp(e - m[dst]); segment sum per (dst, head); ew overwritten with w
__global__ void edge_exp_sum(const int* __restrict__ dst,
                             float* __restrict__ ew,
                             const unsigned* __restrict__ menc,
                             float* __restrict__ ssum, int E) {
  int e = blockIdx.x * blockDim.x + threadIdx.x;
  if (e >= E) return;
  int d = dst[e];
#pragma unroll
  for (int h = 0; h < HEADS; ++h) {
    float m = fdec(menc[d * HEADS + h]);
    if (!__builtin_isfinite(m)) m = 0.0f;  // nodes with no in-edges
    float w = __expf(ew[(size_t)e * HEADS + h] - m);
    ew[(size_t)e * HEADS + h] = w;
    atomicAdd(&ssum[d * HEADS + h], w);
  }
}

// out[dst][c] += H[src][c] * w/(s[dst]+eps)   (one thread per edge-channel)
__global__ void edge_aggregate(const int* __restrict__ src,
                               const int* __restrict__ dst,
                               const float* __restrict__ H,
                               const float* __restrict__ ew,
                               const float* __restrict__ ssum,
                               float* __restrict__ out, long long total) {
  long long i = (long long)blockIdx.x * blockDim.x + threadIdx.x;
  if (i >= total) return;
  int e = (int)(i >> 7);
  int c = (int)(i & 127);
  int h = c >> 5;
  int s = src[e], d = dst[e];
  float alpha = ew[(size_t)e * HEADS + h] / (ssum[d * HEADS + h] + 1e-16f);
  atomicAdd(&out[(size_t)d * HID + c], H[(size_t)s * HID + c] * alpha);
}

// per-channel sum / sum-of-squares of (X + bias), LDS-reduced per block
__global__ void bn_stats(const float* __restrict__ X, const float* __restrict__ b,
                         float* __restrict__ sums, float* __restrict__ sqs, int N) {
  __shared__ float ls[HID], lq[HID];
  int tid = threadIdx.x;
  if (tid < HID) { ls[tid] = 0.f; lq[tid] = 0.f; }
  __syncthreads();
  int c = tid & 127;
  int rh = tid >> 7;  // 0..1
  float bc = b[c];
  int row0 = blockIdx.x * 256 + rh;
  int rowend = blockIdx.x * 256 + 256;
  if (rowend > N) rowend = N;
  float s = 0.f, q = 0.f;
  for (int n = row0; n < rowend; n += 2) {
    float v = X[(size_t)n * HID + c] + bc;
    s += v;
    q += v * v;
  }
  atomicAdd(&ls[c], s);
  atomicAdd(&lq[c], q);
  __syncthreads();
  if (tid < HID) {
    atomicAdd(&sums[tid], ls[tid]);
    atomicAdd(&sqs[tid], lq[tid]);
  }
}

// x = relu(gamma*((x+b)-mu)*rsqrt(var+eps)+beta), in place
__global__ void bn_apply_relu(float* __restrict__ X, const float* __restrict__ b,
                              const float* __restrict__ sums,
                              const float* __restrict__ sqs,
                              const float* __restrict__ gamma,
                              const float* __restrict__ beta, int N) {
  long long i = (long long)blockIdx.x * blockDim.x + threadIdx.x;
  if (i >= (long long)N * HID) return;
  int c = (int)(i & 127);
  float inv_n = 1.0f / (float)N;
  float mu = sums[c] * inv_n;
  float var = sqs[c] * inv_n - mu * mu;
  float v = X[i] + b[c];
  v = gamma[c] * (v - mu) * rsqrtf(var + 1e-5f) + beta[c];
  X[i] = (v > 0.f) ? v : 0.f;
}

// global mean pool (sum + count via atomics)
__global__ void pool_sum(const float* __restrict__ X, const int* __restrict__ batch,
                         float* __restrict__ gsum, float* __restrict__ gcnt, int N) {
  long long i = (long long)blockIdx.x * blockDim.x + threadIdx.x;
  if (i >= (long long)N * HID) return;
  int n = (int)(i >> 7);
  int c = (int)(i & 127);
  int g = batch[n];
  atomicAdd(&gsum[g * HID + c], X[i]);
  if (c == 0) atomicAdd(&gcnt[g], 1.0f);
}

// whole dense head in one workgroup: 128->64 (BN,relu) ->32 (BN,relu) ->4
__global__ void head_mlp(const float* __restrict__ gsum, const float* __restrict__ gcnt,
                         const float* __restrict__ d1W, const float* __restrict__ d1b,
                         const float* __restrict__ g1, const float* __restrict__ be1,
                         const float* __restrict__ d2W, const float* __restrict__ d2b,
                         const float* __restrict__ g2, const float* __restrict__ be2,
                         const float* __restrict__ fcW, const float* __restrict__ fcb,
                         float* __restrict__ out) {
  __shared__ float G[N_GRAPHS * HID];   // 32 KB
  __shared__ float T1[N_GRAPHS * 64];   // 16 KB
  __shared__ float T2[N_GRAPHS * 32];   //  8 KB
  __shared__ float mu[64], va[64];
  int tid = threadIdx.x;

  for (int i = tid; i < N_GRAPHS * HID; i += 256) {
    int g = i >> 7;
    float c = gcnt[g];
    G[i] = gsum[i] / fmaxf(c, 1.0f);
  }
  __syncthreads();

  // T1 = G @ d1W + d1b
  for (int i = tid; i < N_GRAPHS * 64; i += 256) {
    int r = i >> 6, j = i & 63;
    float acc = d1b[j];
    for (int k = 0; k < HID; ++k) acc += G[r * HID + k] * d1W[k * 64 + j];
    T1[i] = acc;
  }
  __syncthreads();
  if (tid < 64) {
    float s = 0.f, q = 0.f;
    for (int r = 0; r < N_GRAPHS; ++r) { float v = T1[r * 64 + tid]; s += v; q += v * v; }
    float m = s * (1.0f / N_GRAPHS);
    mu[tid] = m;
    va[tid] = q * (1.0f / N_GRAPHS) - m * m;
  }
  __syncthreads();
  for (int i = tid; i < N_GRAPHS * 64; i += 256) {
    int j = i & 63;
    float v = g1[j] * (T1[i] - mu[j]) * rsqrtf(va[j] + 1e-5f) + be1[j];
    T1[i] = (v > 0.f) ? v : 0.f;
  }
  __syncthreads();

  // T2 = T1 @ d2W + d2b
  for (int i = tid; i < N_GRAPHS * 32; i += 256) {
    int r = i >> 5, j = i & 31;
    float acc = d2b[j];
    for (int k = 0; k < 64; ++k) acc += T1[r * 64 + k] * d2W[k * 32 + j];
    T2[i] = acc;
  }
  __syncthreads();
  if (tid < 32) {
    float s = 0.f, q = 0.f;
    for (int r = 0; r < N_GRAPHS; ++r) { float v = T2[r * 32 + tid]; s += v; q += v * v; }
    float m = s * (1.0f / N_GRAPHS);
    mu[tid] = m;
    va[tid] = q * (1.0f / N_GRAPHS) - m * m;
  }
  __syncthreads();
  for (int i = tid; i < N_GRAPHS * 32; i += 256) {
    int j = i & 31;
    float v = g2[j] * (T2[i] - mu[j]) * rsqrtf(va[j] + 1e-5f) + be2[j];
    T2[i] = (v > 0.f) ? v : 0.f;
  }
  __syncthreads();

  // out = T2 @ fcW + fcb  (64 x 4 = 256 outputs, one per thread)
  int r = tid >> 2, j = tid & 3;
  float acc = fcb[j];
  for (int k = 0; k < 32; ++k) acc += T2[r * 32 + k] * fcW[k * 4 + j];
  out[tid] = acc;
}

// ---------------------------------------------------------------------------
extern "C" void kernel_launch(void* const* d_in, const int* in_sizes, int n_in,
                              void* d_out, int out_size, void* d_ws, size_t ws_size,
                              hipStream_t stream) {
  const float* nf    = (const float*)d_in[0];
  const int*   ei    = (const int*)d_in[1];
  const int*   batch = (const int*)d_in[2];
  const float* W1    = (const float*)d_in[3];
  const float* b1    = (const float*)d_in[4];
  const float* as1   = (const float*)d_in[5];
  const float* ad1   = (const float*)d_in[6];
  const float* bg1   = (const float*)d_in[7];
  const float* bb1   = (const float*)d_in[8];
  const float* W2    = (const float*)d_in[9];
  const float* b2    = (const float*)d_in[10];
  const float* as2   = (const float*)d_in[11];
  const float* ad2   = (const float*)d_in[12];
  const float* bg2   = (const float*)d_in[13];
  const float* bb2   = (const float*)d_in[14];
  const float* d1W   = (const float*)d_in[15];
  const float* d1b   = (const float*)d_in[16];
  const float* gD1   = (const float*)d_in[17];
  const float* bD1   = (const float*)d_in[18];
  const float* d2W   = (const float*)d_in[19];
  const float* d2b   = (const float*)d_in[20];
  const float* gD2   = (const float*)d_in[21];
  const float* bD2   = (const float*)d_in[22];
  const float* fcW   = (const float*)d_in[23];
  const float* fcb   = (const float*)d_in[24];
  float* out = (float*)d_out;

  const int N = in_sizes[0] / 16;
  const int E = in_sizes[1] / 2;
  const int* src = ei;
  const int* dst = ei + E;

  // bump-allocate workspace (~135 MB)
  char* wsb = (char*)d_ws;
  size_t off = 0;
  auto alloc = [&](size_t bytes) -> void* {
    void* p = wsb + off;
    off = (off + bytes + 255) & ~(size_t)255;
    return p;
  };
  float*    H    = (float*)alloc((size_t)N * HID * 4);
  float*    X2   = (float*)alloc((size_t)N * HID * 4);
  float*    ew   = (float*)alloc((size_t)E * HEADS * 4);
  float*    a_s  = (float*)alloc((size_t)N * HEADS * 4);
  float*    a_d  = (float*)alloc((size_t)N * HEADS * 4);
  unsigned* menc = (unsigned*)alloc((size_t)N * HEADS * 4);
  float*    ssum = (float*)alloc((size_t)N * HEADS * 4);
  float*    bnsum = (float*)alloc(HID * 4);
  float*    bnsq  = (float*)alloc(HID * 4);
  float*    gsum  = (float*)alloc(N_GRAPHS * HID * 4);
  float*    gcnt  = (float*)alloc(N_GRAPHS * 4);

  const int gemm_blocks = (N + 127) / 128;
  const int attn_blocks = (N * HEADS + 255) / 256;
  const int edge_blocks = (E + 255) / 256;
  const long long agg_total = (long long)E * HID;
  const int agg_blocks = (int)((agg_total + 255) / 256);
  const long long nh_total = (long long)N * HID;
  const int nh_blocks = (int)((nh_total + 255) / 256);
  const int stat_blocks = (N + 255) / 256;

  // ----------------- layer 1 -----------------
  gat_linear_wmma<16><<<gemm_blocks, 256, 0, stream>>>(nf, W1, H, N);
  attn_scores<<<attn_blocks, 256, 0, stream>>>(H, as1, ad1, a_s, a_d, N);
  (void)hipMemsetAsync(menc, 0, (size_t)N * HEADS * 4, stream);
  (void)hipMemsetAsync(ssum, 0, (size_t)N * HEADS * 4, stream);
  edge_scores_max<<<edge_blocks, 256, 0, stream>>>(src, dst, a_s, a_d, ew, menc, E);
  edge_exp_sum<<<edge_blocks, 256, 0, stream>>>(dst, ew, menc, ssum, E);
  (void)hipMemsetAsync(X2, 0, (size_t)N * HID * 4, stream);
  edge_aggregate<<<agg_blocks, 256, 0, stream>>>(src, dst, H, ew, ssum, X2, agg_total);
  (void)hipMemsetAsync(bnsum, 0, HID * 4, stream);
  (void)hipMemsetAsync(bnsq, 0, HID * 4, stream);
  bn_stats<<<stat_blocks, 256, 0, stream>>>(X2, b1, bnsum, bnsq, N);
  bn_apply_relu<<<nh_blocks, 256, 0, stream>>>(X2, b1, bnsum, bnsq, bg1, bb1, N);

  // ----------------- layer 2 -----------------
  gat_linear_wmma<128><<<gemm_blocks, 256, 0, stream>>>(X2, W2, H, N);
  attn_scores<<<attn_blocks, 256, 0, stream>>>(H, as2, ad2, a_s, a_d, N);
  (void)hipMemsetAsync(menc, 0, (size_t)N * HEADS * 4, stream);
  (void)hipMemsetAsync(ssum, 0, (size_t)N * HEADS * 4, stream);
  edge_scores_max<<<edge_blocks, 256, 0, stream>>>(src, dst, a_s, a_d, ew, menc, E);
  edge_exp_sum<<<edge_blocks, 256, 0, stream>>>(dst, ew, menc, ssum, E);
  (void)hipMemsetAsync(X2, 0, (size_t)N * HID * 4, stream);  // consumed by GEMM above
  edge_aggregate<<<agg_blocks, 256, 0, stream>>>(src, dst, H, ew, ssum, X2, agg_total);
  (void)hipMemsetAsync(bnsum, 0, HID * 4, stream);
  (void)hipMemsetAsync(bnsq, 0, HID * 4, stream);
  bn_stats<<<stat_blocks, 256, 0, stream>>>(X2, b2, bnsum, bnsq, N);
  bn_apply_relu<<<nh_blocks, 256, 0, stream>>>(X2, b2, bnsum, bnsq, bg2, bb2, N);

  // ----------------- pool + head -----------------
  (void)hipMemsetAsync(gsum, 0, N_GRAPHS * HID * 4, stream);
  (void)hipMemsetAsync(gcnt, 0, N_GRAPHS * 4, stream);
  pool_sum<<<nh_blocks, 256, 0, stream>>>(X2, batch, gsum, gcnt, N);
  head_mlp<<<1, 256, 0, stream>>>(gsum, gcnt, d1W, d1b, gD1, bD1,
                                  d2W, d2b, gD2, bD2, fcW, fcb, out);
}
